// AttentionBlock_49194555408461
// MI455X (gfx1250) — matrix operand order repaired and checked by reading, not compiled, not run on hardware
//
#include <hip/hip_runtime.h>
#include <hip/hip_bf16.h>

typedef _Float16 h16;
typedef __attribute__((ext_vector_type(8)))  _Float16 v8h;
typedef __attribute__((ext_vector_type(16))) _Float16 v16h;
typedef __attribute__((ext_vector_type(8)))  float    v8f;

// ---------------------------------------------------------------------------
// WMMA helper (CDNA5 wave32, 16x16x32 f16 -> f32)
// ---------------------------------------------------------------------------
__device__ __forceinline__ v8f wmma_f16(v16h a, v16h b, v8f c) {
    return __builtin_amdgcn_wmma_f32_16x16x32_f16(
        /*neg_a=*/false, a, /*neg_b=*/false, b,
        /*c_mod=*/(short)0, c, /*reuse_a=*/false, /*reuse_b=*/false);
}

// A fragment 16x32 (MxK): lane m=lane&15, h=lane>>4; halves 0-7 hold K=h*8+0..7,
// halves 8-15 hold K=16+h*8+0..7 (CDNA5 ISA 16-bit A layout).
__device__ __forceinline__ v16h load_a16x32(const h16* p0, int ld) {
    int lane = threadIdx.x & 31;
    int m = lane & 15, hh = (lane >> 4) & 1;
    const h16* p = p0 + (size_t)m * ld + hh * 8;
    v8h lo = *(const v8h*)(p);
    v8h hi = *(const v8h*)(p + 16);
    return __builtin_shufflevector(lo, hi, 0,1,2,3,4,5,6,7,8,9,10,11,12,13,14,15);
}

// B fragment 32x16 (KxN) with B[k][n] = src[n*ld + k] (src row-major [N][K]).
// Lane n=lane&15, h=lane>>4: halves 0-7 hold K=h*8+0..7, halves 8-15 hold K=16+h*8+0..7.
__device__ __forceinline__ v16h load_b32x16(const h16* p0, int ld) {
    int lane = threadIdx.x & 31;
    int n = lane & 15, kh = (lane >> 4) & 1;
    const h16* p = p0 + (size_t)n * ld + kh * 8;
    v8h lo = *(const v8h*)(p);
    v8h hi = *(const v8h*)(p + 16);
    return __builtin_shufflevector(lo, hi, 0,1,2,3,4,5,6,7,8,9,10,11,12,13,14,15);
}

// ---------------------------------------------------------------------------
// Kernel 1: f32 -> f16 convert (weights)
// ---------------------------------------------------------------------------
__global__ void cvt_kernel(const float* __restrict__ src, h16* __restrict__ dst, int n) {
    int i = blockIdx.x * 256 + threadIdx.x;
    if (i < n) dst[i] = (h16)src[i];
}

// ---------------------------------------------------------------------------
// Kernel 2: GroupNorm [B,512,32,32] -> f16 [B, T=1024, C=512]
// one block per (b, group); 16 channels * 1024 spatial contiguous in src.
// ---------------------------------------------------------------------------
__global__ void gn_kernel(const float* __restrict__ x, const float* __restrict__ g,
                          const float* __restrict__ be, h16* __restrict__ xh) {
    int b = blockIdx.x >> 5, grp = blockIdx.x & 31;
    int c0 = grp * 16;
    const float* base = x + ((size_t)b * 512 + c0) * 1024;
    const float4* base4 = (const float4*)base;
    int tid = threadIdx.x;
    float s = 0.f, s2 = 0.f;
    for (int i = tid; i < 4096; i += 256) {
        float4 v = base4[i];
        s  += v.x + v.y + v.z + v.w;
        s2 += v.x * v.x + v.y * v.y + v.z * v.z + v.w * v.w;
    }
    __shared__ float rs[256], rq[256];
    rs[tid] = s; rq[tid] = s2;
    __syncthreads();
    for (int off = 128; off; off >>= 1) {
        if (tid < off) { rs[tid] += rs[tid + off]; rq[tid] += rq[tid + off]; }
        __syncthreads();
    }
    float mu  = rs[0] * (1.f / 16384.f);
    float var = rq[0] * (1.f / 16384.f) - mu * mu;
    float inv = rsqrtf(var + 1e-5f);
    for (int i = tid; i < 4096; i += 256) {
        int e  = i * 4;
        int ch = e >> 10, sp = e & 1023;
        int c  = c0 + ch;
        float ga = g[c] * inv, bb = be[c];
        float4 v = base4[i];
        h16* dst = xh + ((size_t)b * 1024 + sp) * 512 + c;
        dst[0 * 512] = (h16)((v.x - mu) * ga + bb);
        dst[1 * 512] = (h16)((v.y - mu) * ga + bb);
        dst[2 * 512] = (h16)((v.z - mu) * ga + bb);
        dst[3 * 512] = (h16)((v.w - mu) * ga + bb);
    }
}

// ---------------------------------------------------------------------------
// Kernel 3: QKV projection GEMM: Y[8192,512] = X[8192,512] * W^T + bias
// one wave per 32x64 tile, double-buffered fragments (8 WMMAs per k-step).
// mode 0=Q(scale, [B,h,T,d]) 1=K([B,h,T,d]) 2=V([B,h,d,T])
// ---------------------------------------------------------------------------
__global__ void qkv_gemm(const h16* __restrict__ xh, const h16* __restrict__ wh,
                         const float* __restrict__ bias, h16* __restrict__ dst, int mode) {
    int w  = threadIdx.x >> 5;
    int gw = blockIdx.x * 8 + w;            // 256 blocks * 8 waves = 2048 tiles
    int mt = gw >> 3;                       // 0..255 (32-row tiles)
    int n0 = (gw & 7) * 64;
    const h16* arow0 = xh + (size_t)mt * 32 * 512;
    const h16* arow1 = arow0 + 16 * 512;
    v8f acc[8] = {};
    v16h a0 = load_a16x32(arow0, 512);
    v16h a1 = load_a16x32(arow1, 512);
    v16h bf[4];
#pragma unroll
    for (int j = 0; j < 4; ++j)
        bf[j] = load_b32x16(wh + (size_t)(n0 + j * 16) * 512, 512);
    for (int k0 = 0; k0 < 512; k0 += 32) {
        int k1 = (k0 + 32) & 511;           // wrap on last iter (harmless reload)
        v16h na0 = load_a16x32(arow0 + k1, 512);
        v16h na1 = load_a16x32(arow1 + k1, 512);
        v16h nb[4];
#pragma unroll
        for (int j = 0; j < 4; ++j)
            nb[j] = load_b32x16(wh + (size_t)(n0 + j * 16) * 512 + k1, 512);
#pragma unroll
        for (int j = 0; j < 4; ++j) {
            acc[j]     = wmma_f16(a0, bf[j], acc[j]);
            acc[4 + j] = wmma_f16(a1, bf[j], acc[4 + j]);
        }
        a0 = na0; a1 = na1;
#pragma unroll
        for (int j = 0; j < 4; ++j) bf[j] = nb[j];
    }
    int lane = threadIdx.x & 31;
    int nlo = lane & 15, mhi = (lane >> 4) * 8;
    const float qscale = 0.35355339059327373f;  // 64^-0.25 (applied to q only)
#pragma unroll
    for (int half = 0; half < 2; ++half) {
#pragma unroll
        for (int j = 0; j < 4; ++j) {
            int c = n0 + j * 16 + nlo;
            float bv = bias[c];
            int hd = c >> 6, d = c & 63;
#pragma unroll
            for (int r = 0; r < 8; ++r) {
                int gm = mt * 32 + half * 16 + mhi + r;
                int bb = gm >> 10, t = gm & 1023;
                float v = acc[half * 4 + j][r] + bv;
                if (mode == 0) v *= qscale;
                size_t idx;
                if (mode == 2) idx = (((size_t)bb * 8 + hd) * 64 + d) * 1024 + t;  // vT [B,h,d,T]
                else           idx = (((size_t)bb * 8 + hd) * 1024 + t) * 64 + d;  // [B,h,T,d]
                dst[idx] = (h16)v;
            }
        }
    }
}

// ---------------------------------------------------------------------------
// Kernel 4: flash attention. One wave per (b,h, 16-query tile).
// K-fragments double-buffered; V-fragments issued before the softmax VALU
// section so loads overlap the 24 v_exp_f32 + shuffle reductions per tile.
// q,k: [B,h,T,64] f16 (q pre-scaled); vT: [B,h,64,T] f16; ctx -> [B,T,512] f16
// ---------------------------------------------------------------------------
__global__ void attn_kernel(const h16* __restrict__ qh, const h16* __restrict__ kh,
                            const h16* __restrict__ vT, h16* __restrict__ ctx) {
    __shared__ __align__(16) h16 Pt[16 * 32];
    int bh = blockIdx.x >> 6;
    int t0 = (blockIdx.x & 63) * 16;
    int lane = threadIdx.x & 31;
    const h16* qbase = qh + ((size_t)bh * 1024 + t0) * 64;
    const h16* kbase = kh + (size_t)bh * 1024 * 64;
    const h16* vbase = vT + (size_t)bh * 64 * 1024;

    v16h qa0 = load_a16x32(qbase + 0, 64);
    v16h qa1 = load_a16x32(qbase + 32, 64);
    v8f o[4] = {};
    float mrow[8], lrow[8];
#pragma unroll
    for (int r = 0; r < 8; ++r) { mrow[r] = -1e30f; lrow[r] = 0.f; }

    // preload K fragments for j0 = 0
    v16h kb0 = load_b32x16(kbase + 0,            64);
    v16h kb1 = load_b32x16(kbase + 16 * 64,      64);
    v16h kb2 = load_b32x16(kbase + 32,           64);
    v16h kb3 = load_b32x16(kbase + 16 * 64 + 32, 64);

    for (int j0 = 0; j0 < 1024; j0 += 32) {
        // scores S[16 x 32] = Q[16x64] . K^T
        v8f s0 = {}, s1 = {};
        s0 = wmma_f16(qa0, kb0, s0);
        s1 = wmma_f16(qa0, kb1, s1);
        s0 = wmma_f16(qa1, kb2, s0);
        s1 = wmma_f16(qa1, kb3, s1);
        // issue V loads for this tile and K loads for the next tile NOW;
        // the softmax section below hides their latency.
        v16h vb[4];
#pragma unroll
        for (int dt = 0; dt < 4; ++dt)
            vb[dt] = load_b32x16(vbase + (size_t)(dt * 16) * 1024 + j0, 1024);
        int jn = (j0 + 32) & 1023;
        v16h nk0 = load_b32x16(kbase + (size_t)jn * 64,             64);
        v16h nk1 = load_b32x16(kbase + (size_t)(jn + 16) * 64,      64);
        v16h nk2 = load_b32x16(kbase + (size_t)jn * 64 + 32,        64);
        v16h nk3 = load_b32x16(kbase + (size_t)(jn + 16) * 64 + 32, 64);
        // online softmax; row m=(lane>>4)*8+r lives in the 16 lanes of its half
#pragma unroll
        for (int r = 0; r < 8; ++r) {
            float a0 = s0[r], a1 = s1[r];
            float mx = fmaxf(a0, a1);
            mx = fmaxf(mx, __shfl_xor(mx, 1, 32));
            mx = fmaxf(mx, __shfl_xor(mx, 2, 32));
            mx = fmaxf(mx, __shfl_xor(mx, 4, 32));
            mx = fmaxf(mx, __shfl_xor(mx, 8, 32));
            float mnew  = fmaxf(mrow[r], mx);
            float alpha = __expf(mrow[r] - mnew);
            float p0 = __expf(a0 - mnew), p1 = __expf(a1 - mnew);
            float ps = p0 + p1;
            ps += __shfl_xor(ps, 1, 32);
            ps += __shfl_xor(ps, 2, 32);
            ps += __shfl_xor(ps, 4, 32);
            ps += __shfl_xor(ps, 8, 32);
            lrow[r] = lrow[r] * alpha + ps;
            mrow[r] = mnew;
#pragma unroll
            for (int dt = 0; dt < 4; ++dt) o[dt][r] *= alpha;
            int m = ((lane >> 4) << 3) + r, n = lane & 15;
            Pt[m * 32 + n]      = (h16)p0;
            Pt[m * 32 + 16 + n] = (h16)p1;
        }
        __syncthreads();
        // ctx += P[16x32] . V[32x64]
        v16h pa = load_a16x32(Pt, 32);
#pragma unroll
        for (int dt = 0; dt < 4; ++dt)
            o[dt] = wmma_f16(pa, vb[dt], o[dt]);
        __syncthreads();
        kb0 = nk0; kb1 = nk1; kb2 = nk2; kb3 = nk3;
    }
    int b = bh >> 3, hd = bh & 7;
#pragma unroll
    for (int r = 0; r < 8; ++r) {
        float inv = 1.f / lrow[r];
        int t = t0 + ((lane >> 4) << 3) + r;
#pragma unroll
        for (int dt = 0; dt < 4; ++dt) {
            int c = hd * 64 + dt * 16 + (lane & 15);
            ctx[((size_t)b * 1024 + t) * 512 + c] = (h16)(o[dt][r] * inv);
        }
    }
}

// ---------------------------------------------------------------------------
// Kernel 5: out-projection GEMM fused with LayerNorm + residual + NCHW store.
// one block (8 waves) per 32-token tile; wave tile 32x64, double-buffered;
// Y tile (32x512 f32 = 64KB) in LDS (<< 320KB/WGP).
// ---------------------------------------------------------------------------
__global__ void oproj_ln_kernel(const h16* __restrict__ ctx, const h16* __restrict__ woh,
                                const float* __restrict__ bo, const float* __restrict__ lng,
                                const float* __restrict__ lnb, const float* __restrict__ resid,
                                float* __restrict__ out) {
    __shared__ __align__(16) float Yt[32 * 512];
    int w = threadIdx.x >> 5, lane = threadIdx.x & 31;
    int mt = blockIdx.x;                    // 0..255 (32-row tiles)
    int n0 = w * 64;
    const h16* arow0 = ctx + (size_t)mt * 32 * 512;
    const h16* arow1 = arow0 + 16 * 512;
    v8f acc[8] = {};
    v16h a0 = load_a16x32(arow0, 512);
    v16h a1 = load_a16x32(arow1, 512);
    v16h bf[4];
#pragma unroll
    for (int j = 0; j < 4; ++j)
        bf[j] = load_b32x16(woh + (size_t)(n0 + j * 16) * 512, 512);
    for (int k0 = 0; k0 < 512; k0 += 32) {
        int k1 = (k0 + 32) & 511;
        v16h na0 = load_a16x32(arow0 + k1, 512);
        v16h na1 = load_a16x32(arow1 + k1, 512);
        v16h nb[4];
#pragma unroll
        for (int j = 0; j < 4; ++j)
            nb[j] = load_b32x16(woh + (size_t)(n0 + j * 16) * 512 + k1, 512);
#pragma unroll
        for (int j = 0; j < 4; ++j) {
            acc[j]     = wmma_f16(a0, bf[j], acc[j]);
            acc[4 + j] = wmma_f16(a1, bf[j], acc[4 + j]);
        }
        a0 = na0; a1 = na1;
#pragma unroll
        for (int j = 0; j < 4; ++j) bf[j] = nb[j];
    }
    int nlo = lane & 15, mhi = (lane >> 4) * 8;
#pragma unroll
    for (int half = 0; half < 2; ++half) {
#pragma unroll
        for (int j = 0; j < 4; ++j) {
            int c = n0 + j * 16 + nlo;
            float bv = bo[c];
#pragma unroll
            for (int r = 0; r < 8; ++r)
                Yt[(half * 16 + mhi + r) * 512 + c] = acc[half * 4 + j][r] + bv;
        }
    }
    __syncthreads();
    // LayerNorm over C=512 per row; wave w handles rows 4w .. 4w+3
#pragma unroll
    for (int rr = 0; rr < 4; ++rr) {
        int row = w * 4 + rr;
        float s = 0.f, s2 = 0.f;
#pragma unroll
        for (int i = 0; i < 16; ++i) {
            float v = Yt[row * 512 + lane * 16 + i];
            s += v; s2 += v * v;
        }
#pragma unroll
        for (int msk = 16; msk; msk >>= 1) {
            s  += __shfl_xor(s,  msk, 32);
            s2 += __shfl_xor(s2, msk, 32);
        }
        float mu  = s * (1.f / 512.f);
        float var = s2 * (1.f / 512.f) - mu * mu;
        float inv = rsqrtf(var + 1e-5f);
        int gm = mt * 32 + row, b = gm >> 10, t = gm & 1023;
#pragma unroll
        for (int i = 0; i < 16; ++i) {
            int c = lane * 16 + i;
            float v = (Yt[row * 512 + c] - mu) * inv * lng[c] + lnb[c];
            size_t idx = ((size_t)b * 512 + c) * 1024 + t;
            out[idx] = v + resid[idx];  // RESCALE = 1.0
        }
    }
}

// ---------------------------------------------------------------------------
extern "C" void kernel_launch(void* const* d_in, const int* in_sizes, int n_in,
                              void* d_out, int out_size, void* d_ws, size_t ws_size,
                              hipStream_t stream) {
    (void)in_sizes; (void)n_in; (void)out_size; (void)ws_size;
    const float* x   = (const float*)d_in[0];
    const float* gng = (const float*)d_in[1];
    const float* gnb = (const float*)d_in[2];
    const float* wq  = (const float*)d_in[3];
    const float* bq  = (const float*)d_in[4];
    const float* wk  = (const float*)d_in[5];
    const float* bk  = (const float*)d_in[6];
    const float* wv  = (const float*)d_in[7];
    const float* bv  = (const float*)d_in[8];
    const float* wo  = (const float*)d_in[9];
    const float* bo  = (const float*)d_in[10];
    const float* lng = (const float*)d_in[11];
    const float* lnb = (const float*)d_in[12];
    float* out = (float*)d_out;

    char* ws = (char*)d_ws;
    h16* whq = (h16*)(ws);                 // 4 x 512*512 f16 weights = 2 MB
    h16* whk = whq + 262144;
    h16* whv = whk + 262144;
    h16* who = whv + 262144;
    h16* xh  = (h16*)(ws + ((size_t)2  << 20));   // normalized x [8192,512] f16, 8 MB
    h16* qh  = (h16*)(ws + ((size_t)10 << 20));   // q [B,h,T,d] f16, 8 MB
    h16* kh  = (h16*)(ws + ((size_t)18 << 20));   // k [B,h,T,d] f16, 8 MB
    h16* vT  = (h16*)(ws + ((size_t)26 << 20));   // v [B,h,d,T] f16, 8 MB
    h16* ctx = xh;  // xh dead after QKV projections; reuse for ctx (34 MB total)

    cvt_kernel<<<1024, 256, 0, stream>>>(wq, whq, 262144);
    cvt_kernel<<<1024, 256, 0, stream>>>(wk, whk, 262144);
    cvt_kernel<<<1024, 256, 0, stream>>>(wv, whv, 262144);
    cvt_kernel<<<1024, 256, 0, stream>>>(wo, who, 262144);

    gn_kernel<<<256, 256, 0, stream>>>(x, gng, gnb, xh);

    qkv_gemm<<<256, 256, 0, stream>>>(xh, whq, bq, qh, 0);
    qkv_gemm<<<256, 256, 0, stream>>>(xh, whk, bk, kh, 1);
    qkv_gemm<<<256, 256, 0, stream>>>(xh, whv, bv, vT, 2);

    attn_kernel<<<4096, 32, 0, stream>>>(qh, kh, vT, ctx);

    oproj_ln_kernel<<<256, 256, 0, stream>>>(ctx, who, bo, lng, lnb, x, out);
}
